// SATLayer_68143951118413
// MI455X (gfx1250) — matrix operand order repaired and checked by reading, not compiled, not run on hardware
//
#include <hip/hip_runtime.h>
#include <hip/hip_bf16.h>

#define N_NODES 50000
#define N_EDGES 1600000
#define D 128
#define NEG_SLOPE 0.01f

typedef __attribute__((ext_vector_type(16))) _Float16 v16h;
typedef __attribute__((ext_vector_type(8)))  _Float16 v8h;
typedef __attribute__((ext_vector_type(8)))  float    v8f;

// -------- float atomic max via monotonic integer ordering --------
__device__ __forceinline__ void atomicMaxF(float* addr, float v) {
    if (v >= 0.0f) atomicMax((int*)addr, __float_as_int(v));
    else           atomicMin((unsigned int*)addr, __float_as_uint(v));
}

// -------- init: zero out, set segment-max to -inf, denom to 0 --------
__global__ void init_kernel(float* __restrict__ out, float* __restrict__ m,
                            float* __restrict__ denom) {
    int i = blockIdx.x * blockDim.x + threadIdx.x;
    if (i < N_NODES * D) out[i] = 0.0f;
    if (i < N_NODES) { m[i] = -__builtin_inff(); denom[i] = 0.0f; }
}

// -------- projection GEMM: h = X @ W^T + b  (WMMA f16 in / f32 acc) --------
// grid.x = N_NODES/16 row tiles; block = 256 threads = 8 waves, wave w owns
// column tile [16w, 16w+16). W staged once per block into LDS as f16.
__global__ void __launch_bounds__(256)
proj_wmma_kernel(const float* __restrict__ X, const float* __restrict__ W,
                 const float* __restrict__ bias, float* __restrict__ h) {
    __shared__ _Float16 sX[16 * D];     // [m][k], row tile (4 KB)
    __shared__ _Float16 sW[D * D];      // [n][k], full weight (32 KB)

    const int tid = threadIdx.x;
    const int rowBase = blockIdx.x * 16;

    for (int i = tid; i < 16 * D; i += 256)
        sX[i] = (_Float16)X[(size_t)rowBase * D + i];
    for (int i = tid; i < D * D; i += 256)
        sW[i] = (_Float16)W[i];
    __syncthreads();

    const int wave = tid >> 5;
    const int lane = tid & 31;
    const int g    = lane >> 4;    // half-wave group
    const int mn   = lane & 15;    // A-row / B,D-column within tile
    const int colBase = wave * 16;

    v8f c = {};
    #pragma unroll
    for (int k0 = 0; k0 < D; k0 += 32) {
        // A 16x32 f16 layout: halves 0..7  -> K = k0 + 8g + j
        //                     halves 8..15 -> K = k0 + 16 + 8g + j
        const _Float16* pa = &sX[mn * D + k0 + g * 8];
        // B 32x16 f16 layout: halves 0..7 -> K = k0 + 8g + j, halves 8..15 -> K = k0 + 16 + 8g + j
        const _Float16* pb = &sW[(colBase + mn) * D + k0 + g * 8];
        v8h alo = *(const v8h*)(pa);
        v8h ahi = *(const v8h*)(pa + 16);
        v8h blo = *(const v8h*)(pb);
        v8h bhi = *(const v8h*)(pb + 16);
        v16h a, bm;
        #pragma unroll
        for (int j = 0; j < 8; ++j) {
            a[j] = alo[j];  a[8 + j] = ahi[j];
            bm[j] = blo[j]; bm[8 + j] = bhi[j];
        }
        c = __builtin_amdgcn_wmma_f32_16x16x32_f16(
                /*neg_a=*/false, a, /*neg_b=*/false, bm,
                /*c_mod=*/(short)0, c, /*reuse_a=*/false, /*reuse_b=*/false);
    }

    const float bc = bias[colBase + mn];
    // D layout: VGPR r -> M = g*8 + r, N = mn
    #pragma unroll
    for (int r = 0; r < 8; ++r) {
        int row = rowBase + g * 8 + r;
        h[(size_t)row * D + colBase + mn] = c[r] + bc;
    }
}

// -------- per-row attention scores: s1 = h.a1_w + a1_b, s2 likewise --------
// one wave32 per row, lane handles a float4 column slice, shuffle-reduce
__global__ void scores_kernel(const float* __restrict__ h,
                              const float* __restrict__ a1w, const float* __restrict__ a1b,
                              const float* __restrict__ a2w, const float* __restrict__ a2b,
                              float* __restrict__ s1, float* __restrict__ s2) {
    int gid  = blockIdx.x * blockDim.x + threadIdx.x;
    int row  = gid >> 5;
    int lane = gid & 31;
    if (row >= N_NODES) return;
    float4 hv = ((const float4*)(h + (size_t)row * D))[lane];
    float4 w1 = ((const float4*)a1w)[lane];
    float4 w2 = ((const float4*)a2w)[lane];
    float d1 = hv.x * w1.x + hv.y * w1.y + hv.z * w1.z + hv.w * w1.w;
    float d2 = hv.x * w2.x + hv.y * w2.y + hv.z * w2.z + hv.w * w2.w;
    #pragma unroll
    for (int off = 16; off > 0; off >>= 1) {
        d1 += __shfl_xor(d1, off, 32);
        d2 += __shfl_xor(d2, off, 32);
    }
    if (lane == 0) { s1[row] = d1 + a1b[0]; s2[row] = d2 + a2b[0]; }
}

// -------- edge pass 1: e = leakyrelu(s1[src]+s2[dst]); segment max --------
__global__ void edge_max_kernel(const int* __restrict__ edge,
                                const float* __restrict__ s1, const float* __restrict__ s2,
                                float* __restrict__ e_arr, float* __restrict__ m) {
    int i = blockIdx.x * blockDim.x + threadIdx.x;
    if (i >= N_EDGES) return;
    int src = edge[i];
    int dst = edge[N_EDGES + i];
    float e = s1[src] + s2[dst];
    e = (e >= 0.0f) ? e : NEG_SLOPE * e;
    e_arr[i] = e;
    atomicMaxF(&m[src], e);
}

// -------- edge pass 2: ex = exp(e - m[src]); segment sum --------
__global__ void edge_exp_kernel(const int* __restrict__ edge,
                                const float* __restrict__ m,
                                float* __restrict__ e_arr, float* __restrict__ denom) {
    int i = blockIdx.x * blockDim.x + threadIdx.x;
    if (i >= N_EDGES) return;
    int src = edge[i];
    float ex = __expf(e_arr[i] - m[src]);
    e_arr[i] = ex;
    atomicAdd(&denom[src], ex);
}

// -------- edge pass 3: out[src] += (ex/denom[src]) * h[dst] --------
// one wave32 per edge; lane owns a float4 column slice (coalesced 512B gather)
__global__ void scatter_kernel(const int* __restrict__ edge,
                               const float* __restrict__ e_arr,
                               const float* __restrict__ denom,
                               const float* __restrict__ h,
                               float* __restrict__ out) {
    int gid  = blockIdx.x * blockDim.x + threadIdx.x;
    int e    = gid >> 5;
    int lane = gid & 31;
    if (e >= N_EDGES) return;
    int src = edge[e];
    int dst = edge[N_EDGES + e];
    float att = e_arr[e] / denom[src];
    float4 hv = ((const float4*)(h + (size_t)dst * D))[lane];
    float* op = out + (size_t)src * D + lane * 4;
    atomicAdd(op + 0, att * hv.x);
    atomicAdd(op + 1, att * hv.y);
    atomicAdd(op + 2, att * hv.z);
    atomicAdd(op + 3, att * hv.w);
}

extern "C" void kernel_launch(void* const* d_in, const int* in_sizes, int n_in,
                              void* d_out, int out_size, void* d_ws, size_t ws_size,
                              hipStream_t stream) {
    (void)in_sizes; (void)n_in; (void)out_size; (void)ws_size;
    const float* X    = (const float*)d_in[0];
    const int*   edge = (const int*)d_in[1];
    const float* W    = (const float*)d_in[2];
    const float* b    = (const float*)d_in[3];
    const float* a1w  = (const float*)d_in[4];
    const float* a1b  = (const float*)d_in[5];
    const float* a2w  = (const float*)d_in[6];
    const float* a2b  = (const float*)d_in[7];
    float* out = (float*)d_out;

    char* ws = (char*)d_ws;
    float* h     = (float*)ws;  ws += (size_t)N_NODES * D * sizeof(float);
    float* s1    = (float*)ws;  ws += (size_t)N_NODES * sizeof(float);
    float* s2    = (float*)ws;  ws += (size_t)N_NODES * sizeof(float);
    float* m     = (float*)ws;  ws += (size_t)N_NODES * sizeof(float);
    float* denom = (float*)ws;  ws += (size_t)N_NODES * sizeof(float);
    float* e_arr = (float*)ws;  ws += (size_t)N_EDGES * sizeof(float);

    init_kernel<<<(N_NODES * D + 255) / 256, 256, 0, stream>>>(out, m, denom);
    proj_wmma_kernel<<<N_NODES / 16, 256, 0, stream>>>(X, W, b, h);
    scores_kernel<<<(N_NODES * 32 + 255) / 256, 256, 0, stream>>>(h, a1w, a1b, a2w, a2b, s1, s2);
    edge_max_kernel<<<(N_EDGES + 255) / 256, 256, 0, stream>>>(edge, s1, s2, e_arr, m);
    edge_exp_kernel<<<(N_EDGES + 255) / 256, 256, 0, stream>>>(edge, m, e_arr, denom);
    scatter_kernel<<<(N_EDGES * 32 + 255) / 256, 256, 0, stream>>>(edge, e_arr, denom, h, out);
}
